// DecoderMultiHeadAttention_43447889166454
// MI455X (gfx1250) — matrix operand order repaired and checked by reading, not compiled, not run on hardware
//
#include <hip/hip_runtime.h>

// ---------------------------------------------------------------------------
// Types for gfx1250 WMMA (wave32): bf16 A/B fragments (16 halfs = 8 VGPRs),
// f32 C/D accumulator (8 VGPRs).
// ---------------------------------------------------------------------------
typedef __attribute__((ext_vector_type(8)))  __bf16 v8bf;
typedef __attribute__((ext_vector_type(16))) __bf16 v16bf;
typedef __attribute__((ext_vector_type(8)))  float  v8f;
typedef int v4i __attribute__((vector_size(16)));

#define WMMA_BF16(a, b, c) \
  __builtin_amdgcn_wmma_f32_16x16x32_bf16(false, (a), false, (b), (short)0, (c), false, false)

// CDNA5 async global->LDS copy path (ASYNCcnt-tracked), with fallback.
// Probe-discovered prototype: void(v4i AS1*, v4i AS3*, imm int off, imm int cpol)
#if __has_builtin(__builtin_amdgcn_global_load_async_to_lds_b128)
#define HAVE_ASYNC_LDS 1
#endif

#if defined(HAVE_ASYNC_LDS) && __has_builtin(__builtin_amdgcn_s_wait_asynccnt)
#define WAIT_ASYNC(n) __builtin_amdgcn_s_wait_asynccnt(n)
#elif defined(HAVE_ASYNC_LDS)
#define WAIT_ASYNC(n) asm volatile("s_wait_asynccnt " #n ::: "memory")
#else
#define WAIT_ASYNC(n)
#endif

__device__ __forceinline__ void stage16(const __bf16* g, __bf16* l) {
#ifdef HAVE_ASYNC_LDS
  __builtin_amdgcn_global_load_async_to_lds_b128(
      (__attribute__((address_space(1))) v4i*)(void*)g,
      (__attribute__((address_space(3))) v4i*)(void*)l, 0, 0);
#else
  *(v8bf*)l = *(const v8bf*)g;
#endif
}

__device__ __forceinline__ v16bf cat8(v8bf lo, v8bf hi) {
  return __builtin_shufflevector(lo, hi, 0, 1, 2, 3, 4, 5, 6, 7, 8, 9, 10, 11, 12, 13, 14, 15);
}

// Load this lane's A-fragment chunks: 8 bf16 at p, 8 bf16 at p+8 elements.
// (contiguous K order: lanes 0-15 hold K 0..15, lanes 16-31 hold
//  K 16..31 — caller passes p already offset by (lane>>4)*16.)
__device__ __forceinline__ v16bf load_a32(const __bf16* p) {
  v8bf lo = *(const v8bf*)p;
  v8bf hi = *(const v8bf*)(p + 8);
  return cat8(lo, hi);
}

// ---------------------------------------------------------------------------
// f32 -> bf16 conversion (grid-stride)
// ---------------------------------------------------------------------------
__global__ void cvt_f32_bf16(const float* __restrict__ src,
                             unsigned short* __restrict__ dst, long n) {
  long i = (long)blockIdx.x * blockDim.x + threadIdx.x;
  long stride = (long)gridDim.x * blockDim.x;
  for (; i < n; i += stride) {
    __bf16 b = (__bf16)src[i];
    dst[i] = __builtin_bit_cast(unsigned short, b);
  }
}

// ---------------------------------------------------------------------------
// GEMM: C[M,Nn] = A[M,K] @ W[Nn,K]^T   (A, W bf16 row-major, K contiguous)
// Block = 8 waves = 128(M) x 64(N) tile. W tile (64x32 bf16 = 4KB) is staged
// into LDS (double-buffered) with async global->LDS loads; every wave reads
// its B-fragments from LDS (8x weight-traffic reduction), A straight from
// global (rows are private to each wave). K-step = 32.
// mode 0: write f32 row-major to out (final projection -> d_out)
// mode 1: write bf16 row-major (QK buffer)
// mode 2: write bf16 transposed per batch: out[(b*Nn + col)*Sdim + s]
// ---------------------------------------------------------------------------
__global__ __launch_bounds__(256)
void gemm_bf16_wmma(const unsigned short* __restrict__ Au,
                    const unsigned short* __restrict__ Wu,
                    void* __restrict__ outp,
                    int M, int Nn, int K, int mode, int Sdim) {
  const __bf16* A = (const __bf16*)Au;
  const __bf16* W = (const __bf16*)Wu;

  const int tid = threadIdx.x;
  const int lane = tid & 31;
  const int wid = tid >> 5;                 // 0..7
  const int mblocks = M >> 7;               // 128 rows per block
  const int mb = blockIdx.x % mblocks;
  const int nb = blockIdx.x / mblocks;      // 64 cols per block
  const int mbase = (mb << 7) + (wid << 4); // this wave's 16 rows
  const int nbase = nb << 6;
  const int hf  = lane >> 4;                // 0 / 1 (half-wave)
  const int l15 = lane & 15;

  __shared__ __align__(128) __bf16 Wtile[2][64 * 32];

  // Staging map: thread tid copies one 16B chunk: W row = tid>>2, quarter = tid&3.
  const int srow = tid >> 2, sq = tid & 3;
  const __bf16* wsrc = W + (long)(nbase + srow) * K + sq * 8;
  __bf16* wdst[2] = { &Wtile[0][srow * 32 + sq * 8], &Wtile[1][srow * 32 + sq * 8] };

  v8f acc[4];
#pragma unroll
  for (int t = 0; t < 4; ++t)
#pragma unroll
    for (int v = 0; v < 8; ++v) acc[t][v] = 0.0f;

  const __bf16* arow = A + (long)(mbase + l15) * K + hf * 16;

  stage16(wsrc, wdst[0]);                   // prologue: fill buffer 0
  for (int k0 = 0; k0 < K; k0 += 32) {
    const int cur = (k0 >> 5) & 1;
    const bool hasNext = (k0 + 32) < K;
    if (hasNext) {
      stage16(wsrc + k0 + 32, wdst[cur ^ 1]);
      WAIT_ASYNC(1);                        // async loads retire in order
    } else {
      WAIT_ASYNC(0);
    }
    __syncthreads();                        // cross-wave visibility of Wtile[cur]

    v16bf af = load_a32(arow + k0);
#pragma unroll
    for (int t = 0; t < 4; ++t) {
      v16bf bf = *(const v16bf*)&Wtile[cur][(t * 16 + l15) * 32 + hf * 16];
      acc[t] = WMMA_BF16(af, bf, acc[t]);
    }
    __syncthreads();                        // all readers done before overwrite
  }

  // Epilogue. C layout: element (m = v + 8*hf, n = l15) in VGPR v.
  if (mode == 0) {
    float* o = (float*)outp;
#pragma unroll
    for (int t = 0; t < 4; ++t)
#pragma unroll
      for (int v = 0; v < 8; ++v) {
        int row = mbase + v + hf * 8;
        int col = nbase + t * 16 + l15;
        o[(long)row * Nn + col] = acc[t][v];
      }
  } else if (mode == 1) {
    __bf16* o = (__bf16*)outp;
#pragma unroll
    for (int t = 0; t < 4; ++t)
#pragma unroll
      for (int v = 0; v < 8; ++v) {
        int row = mbase + v + hf * 8;
        int col = nbase + t * 16 + l15;
        o[(long)row * Nn + col] = (__bf16)acc[t][v];
      }
  } else {
    // transposed: out[(b*Nn + col)*Sdim + s]; per-lane 8 consecutive s -> 16B store
    __bf16* o = (__bf16*)outp;
    int b = mbase / Sdim;
    int sbase = (mbase % Sdim) + hf * 8;
#pragma unroll
    for (int t = 0; t < 4; ++t) {
      int col = nbase + t * 16 + l15;
      v8bf pk;
#pragma unroll
      for (int v = 0; v < 8; ++v) pk[v] = (__bf16)acc[t][v];
      *(v8bf*)(o + ((long)b * Nn + col) * Sdim + sbase) = pk;
    }
  }
}

// ---------------------------------------------------------------------------
// Flash attention: one wave (32 threads) per (b, h, 16-row q tile).
// QK: bf16 [N*S, 2E] row-major; q at col h*128+d, k at col h*128+64+d.
// Vt: bf16 [N][E][S] (transposed V so P@V B-fragments are contiguous).
// O : bf16 [N*S, E] row-major (feeds final projection as A operand).
// ---------------------------------------------------------------------------
__global__ __launch_bounds__(32)
void attn_wmma(const unsigned short* __restrict__ QKu,
               const unsigned short* __restrict__ Vtu,
               const int* __restrict__ mask,
               unsigned short* __restrict__ Ou) {
  constexpr int S = 2048, E = 1024, HD = 64;
  constexpr float SCALE = 0.125f;          // 1/sqrt(64)
  constexpr float NEG_INF = -1000000.0f;

  const __bf16* QK = (const __bf16*)QKu;
  const __bf16* Vt = (const __bf16*)Vtu;
  __bf16* O = (__bf16*)Ou;

  const int lane = threadIdx.x & 31;
  const int hf = lane >> 4, l15 = lane & 15;
  const int qtile = blockIdx.x & 127;       // S/16 tiles
  const int bh = blockIdx.x >> 7;
  const int b = bh >> 4, h = bh & 15;
  const int qbase = qtile << 4;

  __shared__ __align__(32) __bf16 Pbuf[16 * 32];

  // Q A-fragments (reused across whole K loop): row qbase+l15, d chunks.
  const __bf16* qrow = QK + (long)(b * S + qbase + l15) * (2 * E) + h * (2 * HD);
  v16bf qfrag[2];
#pragma unroll
  for (int ks = 0; ks < 2; ++ks)
    qfrag[ks] = load_a32(qrow + ks * 32 + hf * 16);

  float m_i[8], l_i[8];
#pragma unroll
  for (int v = 0; v < 8; ++v) { m_i[v] = -1.0e30f; l_i[v] = 0.0f; }
  v8f acc[4];
#pragma unroll
  for (int t = 0; t < 4; ++t)
#pragma unroll
    for (int v = 0; v < 8; ++v) acc[t][v] = 0.0f;

  for (int kb = 0; kb < S; kb += 32) {
    // ---- logits: Q(16x64) @ K^T -> two 16x16 tiles (k cols 0..15, 16..31)
    v8f c0, c1;
#pragma unroll
    for (int v = 0; v < 8; ++v) { c0[v] = 0.0f; c1[v] = 0.0f; }
#pragma unroll
    for (int ks = 0; ks < 2; ++ks) {
      const __bf16* kp0 =
          QK + (long)(b * S + kb + l15) * (2 * E) + h * (2 * HD) + HD + ks * 32 + hf * 16;
      v16bf b0 = *(const v16bf*)kp0;
      v16bf b1 = *(const v16bf*)(kp0 + 16L * (2 * E));
      c0 = WMMA_BF16(qfrag[ks], b0, c0);
      c1 = WMMA_BF16(qfrag[ks], b1, c1);
    }

    // ---- mask + scale + online softmax (half-wave shuffles match C layout)
    float p0[8], p1[8], fac[8];
#pragma unroll
    for (int v = 0; v < 8; ++v) {
      int grow = qbase + v + hf * 8;
      int col0 = kb + l15, col1 = kb + 16 + l15;
      float x0 = mask[(long)grow * S + col0] ? c0[v] * SCALE : NEG_INF;
      float x1 = mask[(long)grow * S + col1] ? c1[v] * SCALE : NEG_INF;
      float mx = fmaxf(x0, x1);
      mx = fmaxf(mx, __shfl_xor(mx, 1, 16));
      mx = fmaxf(mx, __shfl_xor(mx, 2, 16));
      mx = fmaxf(mx, __shfl_xor(mx, 4, 16));
      mx = fmaxf(mx, __shfl_xor(mx, 8, 16));
      float mn = fmaxf(m_i[v], mx);
      float f = __expf(m_i[v] - mn);
      m_i[v] = mn;
      p0[v] = __expf(x0 - mn);
      p1[v] = __expf(x1 - mn);
      float rs = p0[v] + p1[v];
      rs += __shfl_xor(rs, 1, 16);
      rs += __shfl_xor(rs, 2, 16);
      rs += __shfl_xor(rs, 4, 16);
      rs += __shfl_xor(rs, 8, 16);
      l_i[v] = l_i[v] * f + rs;
      fac[v] = f;
    }
#pragma unroll
    for (int t = 0; t < 4; ++t)
#pragma unroll
      for (int v = 0; v < 8; ++v) acc[t][v] *= fac[v];

    // ---- P: C layout -> A layout via LDS (16x32 bf16 tile)
#pragma unroll
    for (int v = 0; v < 8; ++v) {
      int rl = v + hf * 8;
      Pbuf[rl * 32 + l15] = (__bf16)p0[v];
      Pbuf[rl * 32 + 16 + l15] = (__bf16)p1[v];
    }
    __syncthreads();  // single-wave block; DS ops are in-order per wave
    v16bf pf = load_a32(&Pbuf[l15 * 32 + hf * 16]);

    // ---- acc += P(16x32) @ V(32x64), 4 d-tiles
#pragma unroll
    for (int t = 0; t < 4; ++t) {
      const __bf16* vp =
          Vt + ((long)b * E + h * HD + t * 16 + l15) * S + kb + hf * 16;
      v16bf bf = *(const v16bf*)vp;
      acc[t] = WMMA_BF16(pf, bf, acc[t]);
    }
    __syncthreads();
  }

  // ---- normalize + store O (row-major token x E)
#pragma unroll
  for (int t = 0; t < 4; ++t)
#pragma unroll
    for (int v = 0; v < 8; ++v) {
      int grow = qbase + v + hf * 8;
      int col = h * HD + t * 16 + l15;
      O[(long)(b * S + grow) * E + col] = (__bf16)(acc[t][v] / l_i[v]);
    }
}

// ---------------------------------------------------------------------------
// Host launcher
// ---------------------------------------------------------------------------
extern "C" void kernel_launch(void* const* d_in, const int* in_sizes, int n_in,
                              void* d_out, int out_size, void* d_ws, size_t ws_size,
                              hipStream_t stream) {
  const float* x_enc = (const float*)d_in[0];  // (2,2048,1024)
  const float* x_dec = (const float*)d_in[1];  // (2,2048,1024)
  const int*   maskp = (const int*)d_in[2];    // (2048,2048)
  const float* W_qk  = (const float*)d_in[3];  // (2048,1024)
  const float* W_v   = (const float*)d_in[4];  // (1024,1024)
  const float* W_o   = (const float*)d_in[5];  // (1024,1024)

  constexpr long N = 2, S = 2048, E = 1024;
  constexpr long M = N * S;  // 4096 token rows

  // Workspace layout (bf16 elements), ~56 MB total, all 256B-aligned.
  unsigned short* Xe  = (unsigned short*)d_ws;
  unsigned short* Xd  = Xe  + (size_t)M * E;            // 8 MB
  unsigned short* Wqk = Xd  + (size_t)M * E;            // 8 MB
  unsigned short* Wvb = Wqk + (size_t)2 * E * E;        // 4 MB
  unsigned short* Wob = Wvb + (size_t)E * E;            // 2 MB
  unsigned short* QKb = Wob + (size_t)E * E;            // 2 MB
  unsigned short* Vtb = QKb + (size_t)M * 2 * E;        // 16 MB
  unsigned short* Ob  = Vtb + (size_t)N * E * S;        // 8 MB
                                                        // Ob: 8 MB

  // 1) convert inputs / weights to bf16
  cvt_f32_bf16<<<2048, 256, 0, stream>>>(x_enc, Xe, M * E);
  cvt_f32_bf16<<<2048, 256, 0, stream>>>(x_dec, Xd, M * E);
  cvt_f32_bf16<<<2048, 256, 0, stream>>>(W_qk, Wqk, 2 * E * E);
  cvt_f32_bf16<<<1024, 256, 0, stream>>>(W_v, Wvb, E * E);
  cvt_f32_bf16<<<1024, 256, 0, stream>>>(W_o, Wob, E * E);

  // 2) QK = Xe @ Wqk^T -> bf16 row-major (4096 x 2048): (32 mb)*(32 nb) blocks
  gemm_bf16_wmma<<<1024, 256, 0, stream>>>(Xe, Wqk, QKb, (int)M, (int)(2 * E),
                                           (int)E, 1, (int)S);
  // 3) V = Xd @ Wv^T -> bf16 transposed [b][E][S]: (32)*(16) = 512 blocks
  gemm_bf16_wmma<<<512, 256, 0, stream>>>(Xd, Wvb, Vtb, (int)M, (int)E,
                                          (int)E, 2, (int)S);
  // 4) attention: (S/16) * N * H = 128 * 32 = 4096 one-wave blocks
  attn_wmma<<<4096, 32, 0, stream>>>(QKb, Vtb, maskp, Ob);
  // 5) out = O @ Wo^T -> f32 (4096 x 1024): 512 blocks
  gemm_bf16_wmma<<<512, 256, 0, stream>>>(Ob, Wob, d_out, (int)M, (int)E,
                                          (int)E, 0, (int)S);
}
